// HGTEncoder_2748779070006
// MI455X (gfx1250) — matrix operand order repaired and checked by reading, not compiled, not run on hardware
//
#include <hip/hip_runtime.h>
#include <hip/hip_bf16.h>

// ---------------------------------------------------------------------------
// HGT encoder for MI455X (gfx1250, wave32).
// Dense GEMMs use v_wmma_f32_16x16x32_bf16 (fp32 accumulate), LDS tiles laid
// out so each WMMA fragment is exactly two ds_load_b128 ops, and global->LDS
// staging is software-pipelined through registers (loads overlap WMMA).
// Relation transforms hoisted from edge-level to node-level batched GEMMs.
// Segment softmax + message scatter via fp32 global atomics.
// ---------------------------------------------------------------------------

typedef __attribute__((ext_vector_type(16))) __bf16 v16bf;
typedef __attribute__((ext_vector_type(8)))  __bf16 bf16x8;
typedef __attribute__((ext_vector_type(4)))  __bf16 bf16x4;
typedef __attribute__((ext_vector_type(8)))  float  v8f;

union Frag16 { v16bf v; bf16x8 h[2]; };

#define HIDDEN 768
#define HEADS  8
#define HD     96     // head dim

// LDS row padding: +8 bf16 keeps every 16B fragment run 16B-aligned and gives
// an 80B (20-dword) row stride -> conflict-free across 16 lanes.
#define APAD 8

// ======================= main GEMM: C = act(A[MxK] * W[KxN] + b) ============
// Block tile 128x64, 256 threads = 8 waves (4x2), wave tile 32x32 = 2x2 WMMA.
// N and K must be multiples of TN/TK (always 768 here).
#define TM 128
#define TN 64
#define TK 32

__global__ __launch_bounds__(256) void gemm_bias_act(
    const float* __restrict__ A, const float* __restrict__ W,
    const float* __restrict__ bias, float* __restrict__ C,
    int M, int N, int K, int act)
{
    __shared__ __bf16 As [TM][TK + APAD];   // row-major K tile of A
    __shared__ __bf16 Bst[TN][TK + APAD];   // B tile TRANSPOSED: [n][k]

    const int tid  = threadIdx.x;
    const int lane = tid & 31;
    const int wave = tid >> 5;
    const int wr   = wave >> 1;   // 0..3
    const int wc   = wave & 1;    // 0..1
    const int m_blk = blockIdx.y * TM;
    const int n_blk = blockIdx.x * TN;

    const int lm = lane & 15;
    const int hi = (lane >> 4) & 1;
    const int kbase = hi * 16;

    v8f acc[2][2] = {};

    float4 aReg[4];   // staged A chunks (4 float4 per thread)
    float4 bReg[2];   // staged B chunks (2 float4 per thread)

    // Branch-free staging: OOB A rows are clamped (their products only feed
    // C rows that are never stored, so stale data is harmless).
    auto loadTiles = [&](int k0) {
        #pragma unroll
        for (int t = 0; t < 4; ++t) {
            int j   = tid + t * 256;
            int row = j >> 3;                 // TK/4 = 8 float4 per row
            int c4  = (j & 7) * 4;
            int gm  = m_blk + row;
            gm = (gm < M) ? gm : (M - 1);     // clamp, no exec-mask games
            const float* ap = &A[(size_t)gm * K + (k0 + c4)];
            aReg[t] = *(const float4*)ap;
            __builtin_prefetch(ap + TK, 0, 3);   // speculative: safe at tail
        }
        #pragma unroll
        for (int t = 0; t < 2; ++t) {
            int j  = tid + t * 256;
            int kr = j >> 4;                  // TN/4 = 16 float4 per k-row
            int n4 = (j & 15) * 4;
            bReg[t] = *(const float4*)&W[(size_t)(k0 + kr) * N + (n_blk + n4)];
        }
    };
    auto storeTiles = [&]() {
        #pragma unroll
        for (int t = 0; t < 4; ++t) {
            int j   = tid + t * 256;
            int row = j >> 3;
            int c4  = (j & 7) * 4;
            bf16x4 b;
            b[0] = (__bf16)aReg[t].x; b[1] = (__bf16)aReg[t].y;
            b[2] = (__bf16)aReg[t].z; b[3] = (__bf16)aReg[t].w;
            *(bf16x4*)&As[row][c4] = b;       // 8B ds_store
        }
        #pragma unroll
        for (int t = 0; t < 2; ++t) {
            int j  = tid + t * 256;
            int kr = j >> 4;
            int n4 = (j & 15) * 4;
            Bst[n4 + 0][kr] = (__bf16)bReg[t].x;
            Bst[n4 + 1][kr] = (__bf16)bReg[t].y;
            Bst[n4 + 2][kr] = (__bf16)bReg[t].z;
            Bst[n4 + 3][kr] = (__bf16)bReg[t].w;
        }
    };

    loadTiles(0);
    for (int k0 = 0; k0 < K; k0 += TK) {
        storeTiles();
        __syncthreads();
        if (k0 + TK < K) loadTiles(k0 + TK);  // overlaps with WMMA below

        #pragma unroll
        for (int i = 0; i < 2; ++i) {
            // A fragment: lane m fixed, k = 16*hi + 0..15 contiguous
            //   lanes 0-15:  k 0..15 ; lanes 16-31: k 16..31
            Frag16 a;
            const int mrow = wr * 32 + i * 16 + lm;
            a.h[0] = *(const bf16x8*)&As[mrow][kbase];
            a.h[1] = *(const bf16x8*)&As[mrow][kbase + 8];
            #pragma unroll
            for (int j = 0; j < 2; ++j) {
                // B fragment: lane n fixed, k = 16*hi + 0..15 contiguous
                Frag16 b;
                const int ncol = wc * 32 + j * 16 + lm;
                b.h[0] = *(const bf16x8*)&Bst[ncol][kbase];
                b.h[1] = *(const bf16x8*)&Bst[ncol][kbase + 8];
                acc[i][j] = __builtin_amdgcn_wmma_f32_16x16x32_bf16(
                    false, a.v, false, b.v, (short)0, acc[i][j], false, false);
            }
        }
        __syncthreads();
    }

    // Store: C/D layout -> VGPR r holds M = r + 8*hi, lane -> N
    #pragma unroll
    for (int i = 0; i < 2; ++i) {
        #pragma unroll
        for (int j = 0; j < 2; ++j) {
            int nc = n_blk + wc * 32 + j * 16 + lm;
            float b = bias ? bias[nc] : 0.0f;
            #pragma unroll
            for (int r = 0; r < 8; ++r) {
                int mr = m_blk + wr * 32 + i * 16 + hi * 8 + r;
                if (mr < M) {
                    float v = acc[i][j][r] + b;
                    if (act == 1) v = fmaxf(v, 0.0f);
                    C[(size_t)mr * N + nc] = v;
                }
            }
        }
    }
}

// ========== per-head relation GEMM: C[n,h*96+f] = sum_d A[n,h*96+d]*W[h,d,f]
// grid = (ceil(N/64), HEADS); block tile 64x96, 8 waves (4x2), wave 16x48.
__global__ __launch_bounds__(256) void head_gemm(
    const float* __restrict__ A, const float* __restrict__ Wrel,
    float* __restrict__ C, int Nnodes)
{
    __shared__ __bf16 As [64][TK + APAD];
    __shared__ __bf16 Bst[HD][TK + APAD];   // transposed: [f][d]

    const int head = blockIdx.y;
    const float* Ah = A + head * HD;          // row stride HIDDEN
    const float* Wh = Wrel + (size_t)head * HD * HD;
    float* Ch = C + head * HD;

    const int tid  = threadIdx.x;
    const int lane = tid & 31;
    const int wave = tid >> 5;
    const int wr   = wave >> 1;   // 0..3 -> 16 rows each
    const int wc   = wave & 1;    // 0..1 -> 48 cols each
    const int m_blk = blockIdx.x * 64;

    const int lm = lane & 15;
    const int hi = (lane >> 4) & 1;
    const int kbase = hi * 16;

    v8f acc[3] = {};

    float4 aReg[2];
    float4 bReg[3];

    auto loadTiles = [&](int k0) {
        #pragma unroll
        for (int t = 0; t < 2; ++t) {          // A tile 64x32: 512 float4
            int j   = tid + t * 256;
            int row = j >> 3;
            int c4  = (j & 7) * 4;
            int gm  = m_blk + row;
            gm = (gm < Nnodes) ? gm : (Nnodes - 1);   // clamp
            aReg[t] = *(const float4*)&Ah[(size_t)gm * HIDDEN + (k0 + c4)];
        }
        #pragma unroll
        for (int t = 0; t < 3; ++t) {          // W tile 32x96: 768 float4
            int j  = tid + t * 256;
            int kr = j / 24;                   // HD/4 = 24 float4 per d-row
            int f4 = (j % 24) * 4;
            bReg[t] = *(const float4*)&Wh[(size_t)(k0 + kr) * HD + f4];
        }
    };
    auto storeTiles = [&]() {
        #pragma unroll
        for (int t = 0; t < 2; ++t) {
            int j   = tid + t * 256;
            int row = j >> 3;
            int c4  = (j & 7) * 4;
            bf16x4 b;
            b[0] = (__bf16)aReg[t].x; b[1] = (__bf16)aReg[t].y;
            b[2] = (__bf16)aReg[t].z; b[3] = (__bf16)aReg[t].w;
            *(bf16x4*)&As[row][c4] = b;
        }
        #pragma unroll
        for (int t = 0; t < 3; ++t) {
            int j  = tid + t * 256;
            int kr = j / 24;
            int f4 = (j % 24) * 4;
            Bst[f4 + 0][kr] = (__bf16)bReg[t].x;
            Bst[f4 + 1][kr] = (__bf16)bReg[t].y;
            Bst[f4 + 2][kr] = (__bf16)bReg[t].z;
            Bst[f4 + 3][kr] = (__bf16)bReg[t].w;
        }
    };

    loadTiles(0);
    for (int k0 = 0; k0 < HD; k0 += TK) {
        storeTiles();
        __syncthreads();
        if (k0 + TK < HD) loadTiles(k0 + TK);

        Frag16 a;
        const int mrow = wr * 16 + lm;
        a.h[0] = *(const bf16x8*)&As[mrow][kbase];
        a.h[1] = *(const bf16x8*)&As[mrow][kbase + 8];
        #pragma unroll
        for (int j = 0; j < 3; ++j) {
            Frag16 b;
            const int ncol = wc * 48 + j * 16 + lm;
            b.h[0] = *(const bf16x8*)&Bst[ncol][kbase];
            b.h[1] = *(const bf16x8*)&Bst[ncol][kbase + 8];
            acc[j] = __builtin_amdgcn_wmma_f32_16x16x32_bf16(
                false, a.v, false, b.v, (short)0, acc[j], false, false);
        }
        __syncthreads();
    }

    #pragma unroll
    for (int j = 0; j < 3; ++j) {
        int lcol = wc * 48 + j * 16 + lm;       // < 96 always
        #pragma unroll
        for (int r = 0; r < 8; ++r) {
            int mr = m_blk + wr * 16 + hi * 8 + r;
            if (mr < Nnodes) Ch[(size_t)mr * HIDDEN + lcol] = acc[j][r];
        }
    }
}

// ============================ edge-phase kernels ============================
__device__ inline void atomicMaxF(float* addr, float val) {
    unsigned int* ua = (unsigned int*)addr;
    unsigned int old = *ua;
    while (true) {
        float f = __uint_as_float(old);
        if (f >= val) break;
        unsigned int assumed = old;
        old = atomicCAS(ua, assumed, __float_as_uint(val));
        if (old == assumed) break;
    }
}

__global__ void init_softmax(float* __restrict__ m, float* __restrict__ d, size_t n) {
    size_t i = (size_t)blockIdx.x * blockDim.x + threadIdx.x;
    if (i < n) { m[i] = -INFINITY; d[i] = 0.0f; }
}

__global__ void zero_buf(float* __restrict__ p, size_t n) {
    size_t i = (size_t)blockIdx.x * blockDim.x + threadIdx.x;
    if (i < n) p[i] = 0.0f;
}

// pass 1: logits[e,h] = <q[dst,h,:], krel[src,h,:]> * p_rel[h] / sqrt(D); seg-max
__global__ void edge_logits(const float* __restrict__ q, const float* __restrict__ krel,
                            const int* __restrict__ src, const int* __restrict__ dst,
                            const float* __restrict__ prel, float scale,
                            float* __restrict__ logits, float* __restrict__ mbuf, int E)
{
    long idx = (long)blockIdx.x * blockDim.x + threadIdx.x;
    if (idx >= (long)E * HEADS) return;
    int e = (int)(idx >> 3), h = (int)(idx & 7);
    int s = src[e], d = dst[e];
    const float4* qp = (const float4*)(q    + (size_t)d * HIDDEN + h * HD);
    const float4* kp = (const float4*)(krel + (size_t)s * HIDDEN + h * HD);
    float acc = 0.0f;
    #pragma unroll
    for (int i = 0; i < HD / 4; ++i) {
        float4 a = qp[i], b = kp[i];
        acc += a.x * b.x + a.y * b.y + a.z * b.z + a.w * b.w;
    }
    float lg = acc * prel[h] * scale;
    logits[idx] = lg;
    atomicMaxF(mbuf + (size_t)d * HEADS + h, lg);
}

// pass 2: logits <- exp(logit - max); seg-sum denominators
__global__ void edge_exp(const int* __restrict__ dst, float* __restrict__ logits,
                         const float* __restrict__ mbuf, float* __restrict__ dbuf, int E)
{
    long idx = (long)blockIdx.x * blockDim.x + threadIdx.x;
    if (idx >= (long)E * HEADS) return;
    int e = (int)(idx >> 3), h = (int)(idx & 7);
    int d = dst[e];
    float a = __expf(logits[idx] - mbuf[(size_t)d * HEADS + h]);
    logits[idx] = a;
    atomicAdd(dbuf + (size_t)d * HEADS + h, a);
}

// pass 3: aggr[dst] += vrel[src] * alpha
__global__ void edge_scatter(const float* __restrict__ vrel, const int* __restrict__ src,
                             const int* __restrict__ dst, const float* __restrict__ logits,
                             const float* __restrict__ dbuf, float* __restrict__ aggr, int E)
{
    long idx = (long)blockIdx.x * blockDim.x + threadIdx.x;
    if (idx >= (long)E * HIDDEN) return;
    int e = (int)(idx / HIDDEN);
    int c = (int)(idx % HIDDEN);
    int h = c / HD;
    int s = src[e], d = dst[e];
    float alpha = logits[(size_t)e * HEADS + h] / (dbuf[(size_t)d * HEADS + h] + 1e-16f);
    atomicAdd(aggr + (size_t)d * HIDDEN + c, vrel[(size_t)s * HIDDEN + c] * alpha);
}

// ============================ elementwise kernels ===========================
__global__ void gelu_kernel(const float* __restrict__ x, float* __restrict__ y, size_t n) {
    size_t i = (size_t)blockIdx.x * blockDim.x + threadIdx.x;
    if (i < n) {
        float v = x[i];
        y[i] = 0.5f * v * (1.0f + erff(v * 0.7071067811865475f));
    }
}

// out = relu(beta*o + (1-beta)*h), beta = sigmoid(skip[0])
__global__ void blend_relu(const float* __restrict__ o, const float* __restrict__ h,
                           const float* __restrict__ skip, float* __restrict__ out, size_t n)
{
    size_t i = (size_t)blockIdx.x * blockDim.x + threadIdx.x;
    if (i >= n) return;
    float beta = 1.0f / (1.0f + __expf(-skip[0]));
    out[i] = fmaxf(beta * o[i] + (1.0f - beta) * h[i], 0.0f);
}

// =============================== host driver ================================
static inline size_t cdiv(size_t a, size_t b) { return (a + b - 1) / b; }

extern "C" void kernel_launch(void* const* d_in, const int* in_sizes, int n_in,
                              void* d_out, int out_size, void* d_ws, size_t ws_size,
                              hipStream_t stream)
{
    const float* x_paper  = (const float*)d_in[0];
    const float* x_author = (const float*)d_in[1];
    const float* nlW      = (const float*)d_in[2];
    const float* nlb      = (const float*)d_in[3];
    const float* kqvaW    = (const float*)d_in[4];
    const float* kqvab    = (const float*)d_in[5];
    const float* skipP    = (const float*)d_in[6];
    const float* aRel     = (const float*)d_in[7];
    const float* mRel     = (const float*)d_in[8];
    const float* pRel     = (const float*)d_in[9];
    const int* esrc[3] = {(const int*)d_in[10], (const int*)d_in[12], (const int*)d_in[14]};
    const int* edst[3] = {(const int*)d_in[11], (const int*)d_in[13], (const int*)d_in[15]};
    const int stype[3] = {0, 1, 0};
    const int dtype[3] = {0, 0, 1};

    const int Np = in_sizes[0] / HIDDEN;
    const int Na = in_sizes[1] / HIDDEN;
    const int E  = in_sizes[10];
    const int Nmax = (Np > Na) ? Np : Na;
    const int Ns[2] = {Np, Na};
    const size_t npf = (size_t)Np * HIDDEN;
    const size_t naf = (size_t)Na * HIDDEN;
    const size_t nmf = (size_t)Nmax * HIDDEN;
    const float scale = 1.0f / sqrtf((float)HD);
    const int L = 2;

    // carve workspace
    float* p = (float*)d_ws;
    auto take = [&](size_t n) { float* r = p; p += n; return r; };
    float* hb[2]; float* kb[2]; float* qb[2]; float* vb[2]; float* ag[2];
    hb[0] = take(npf); hb[1] = take(naf);
    kb[0] = take(npf); qb[0] = take(npf); vb[0] = take(npf);
    kb[1] = take(naf); qb[1] = take(naf); vb[1] = take(naf);
    float* krel   = take(nmf);
    float* vrel   = take(nmf);
    ag[0] = take(npf); ag[1] = take(naf);
    float* tmp    = take(nmf);
    float* logits = take((size_t)E * HEADS);
    float* mbuf   = take((size_t)Nmax * HEADS);
    float* dbuf   = take((size_t)Nmax * HEADS);

    auto gemm = [&](const float* A, const float* W, const float* b, float* C,
                    int M, int act) {
        dim3 g(HIDDEN / TN, (unsigned)cdiv(M, TM));
        gemm_bias_act<<<g, 256, 0, stream>>>(A, W, b, C, M, HIDDEN, HIDDEN, act);
    };
    auto hgemm = [&](const float* A, const float* Wr, float* C, int M) {
        dim3 g((unsigned)cdiv(M, 64), HEADS);
        head_gemm<<<g, 256, 0, stream>>>(A, Wr, C, M);
    };

    // input projections + ReLU
    gemm(x_paper,  nlW,                    nlb,          hb[0], Np, 1);
    gemm(x_author, nlW + HIDDEN * HIDDEN, nlb + HIDDEN, hb[1], Na, 1);

    float* dout = (float*)d_out;

    for (int l = 0; l < L; ++l) {
        zero_buf<<<(unsigned)cdiv(npf, 256), 256, 0, stream>>>(ag[0], npf);
        zero_buf<<<(unsigned)cdiv(naf, 256), 256, 0, stream>>>(ag[1], naf);

        // K/Q/V projections per node type
        for (int t = 0; t < 2; ++t) {
            const float* Wl = kqvaW + ((size_t)(l * 2 + t) * 4) * HIDDEN * HIDDEN;
            const float* bl = kqvab + ((size_t)(l * 2 + t) * 4) * HIDDEN;
            gemm(hb[t], Wl + 0 * (size_t)HIDDEN * HIDDEN, bl + 0 * HIDDEN, kb[t], Ns[t], 0);
            gemm(hb[t], Wl + 1 * (size_t)HIDDEN * HIDDEN, bl + 1 * HIDDEN, qb[t], Ns[t], 0);
            gemm(hb[t], Wl + 2 * (size_t)HIDDEN * HIDDEN, bl + 2 * HIDDEN, vb[t], Ns[t], 0);
        }

        // edge types, sequentially (softmax state reused)
        for (int e = 0; e < 3; ++e) {
            int st = stype[e], dt = dtype[e];
            int Ss = Ns[st], Sd = Ns[dt];
            const float* aR = aRel + (size_t)(l * 3 + e) * HEADS * HD * HD;
            const float* mR = mRel + (size_t)(l * 3 + e) * HEADS * HD * HD;
            const float* pR = pRel + (size_t)(l * 3 + e) * HEADS;

            // node-level relation transforms (hoisted from edges)
            hgemm(kb[st], aR, krel, Ss);
            hgemm(vb[st], mR, vrel, Ss);

            size_t nsd = (size_t)Sd * HEADS;
            init_softmax<<<(unsigned)cdiv(nsd, 256), 256, 0, stream>>>(mbuf, dbuf, nsd);

            long ne8 = (long)E * HEADS;
            edge_logits<<<(unsigned)cdiv(ne8, 256), 256, 0, stream>>>(
                qb[dt], krel, esrc[e], edst[e], pR, scale, logits, mbuf, E);
            edge_exp<<<(unsigned)cdiv(ne8, 256), 256, 0, stream>>>(
                edst[e], logits, mbuf, dbuf, E);
            long neh = (long)E * HIDDEN;
            edge_scatter<<<(unsigned)cdiv(neh, 256), 256, 0, stream>>>(
                vrel, esrc[e], edst[e], logits, dbuf, ag[dt], E);
        }

        // output projection + sigmoid-skip blend + per-layer ReLU
        for (int t = 0; t < 2; ++t) {
            size_t nt = (size_t)Ns[t] * HIDDEN;
            const float* Wl = kqvaW + ((size_t)(l * 2 + t) * 4 + 3) * HIDDEN * HIDDEN;
            const float* bl = kqvab + ((size_t)(l * 2 + t) * 4 + 3) * HIDDEN;
            gelu_kernel<<<(unsigned)cdiv(nt, 256), 256, 0, stream>>>(ag[t], tmp, nt);
            gemm(tmp, Wl, bl, krel /* reuse as 'o' */, Ns[t], 0);
            float* out_t = (l == L - 1) ? (t ? dout + npf : dout) : hb[t];
            blend_relu<<<(unsigned)cdiv(nt, 256), 256, 0, stream>>>(
                krel, hb[t], skipP + l * 2 + t, out_t, nt);
        }
    }
}
